// Almtx_27470610825784
// MI455X (gfx1250) — matrix operand order repaired and checked by reading, not compiled, not run on hardware
//
#include <hip/hip_runtime.h>
#include <hip/hip_bf16.h>
#include <math.h>

// Problem constants (from reference): B=16, L=4096, D=128, K=256.
#define BATCH   16
#define LSEQ    4096
#define DDIM    128
#define KOUT    256      // K-1 = 255 padded to 256
#define WIDTH_F 0.125f
#define MASKV  (-1e30f)

typedef __attribute__((ext_vector_type(16))) __bf16 v16bf;
typedef __attribute__((ext_vector_type(8)))  __bf16 v8bf;
typedef __attribute__((ext_vector_type(8)))  float  v8f;
typedef __attribute__((ext_vector_type(4)))  float  v4f;

__device__ __forceinline__ float fast_tanhf(float x) {
    // tanh(x) = 1 - 2/(exp(2x)+1); exp overflow -> +1, underflow -> -1. 1 trans op.
    return 1.f - 2.f / (__expf(2.f * x) + 1.f);
}

// ---------------------------------------------------------------------------
// Kernel 0: pre-pack W1/W2 into bf16 in per-lane WMMA B-fragment layout.
//   W1p[((kc/32)*16 + nt)*32 + lane][0..15]  (4 kchunks)   = 64 KB
//   W2p[((kc/32)*16 + nt)*32 + lane][0..15]  (8 kchunks)   = 128 KB
// Fragment element i for lane: K = kc + (lane>>4)*8 + i + (i&8), N = nt*16 + (lane&15).
// The n==255 zero-pad guard lives here, not in the hot GEMM loop.
// ---------------------------------------------------------------------------
__global__ __launch_bounds__(256) void pack_weights_kernel(const float* __restrict__ W1,
                                                           const float* __restrict__ W2,
                                                           __bf16* __restrict__ W1p,
                                                           __bf16* __restrict__ W2p) {
    const int fid = blockIdx.x * 256 + threadIdx.x;   // fragment-lane id
    if (fid < 2048) {                                 // W1: 4*16*32 fragment-lanes
        const int lane = fid & 31, nt = (fid >> 5) & 15, kc = (fid >> 9) * 32;
        const int n  = nt * 16 + (lane & 15);
        const int kb = kc + (lane >> 4) * 8;
        __bf16* dst = W1p + (size_t)fid * 16;
        #pragma unroll
        for (int i = 0; i < 16; ++i)
            dst[i] = (__bf16)W1[(size_t)(kb + i + (i & 8)) * KOUT + n];
    } else if (fid < 2048 + 4096) {                   // W2: 8*16*32 fragment-lanes
        const int f = fid - 2048;
        const int lane = f & 31, nt = (f >> 5) & 15, kc = (f >> 9) * 32;
        const int n  = nt * 16 + (lane & 15);
        const int kb = kc + (lane >> 4) * 8;
        __bf16* dst = W2p + (size_t)f * 16;
        #pragma unroll
        for (int i = 0; i < 16; ++i)
            dst[i] = (n < KOUT - 1) ? (__bf16)W2[(size_t)(kb + i + (i & 8)) * (KOUT - 1) + n]
                                    : (__bf16)0.f;
    }
}

// ---------------------------------------------------------------------------
// Kernel 1: attnT[b,k,l] = mask-combine( tanh(X@W1+b1)@W2 ) via bf16 WMMA,
// with r[b,l] = X[b,l,:].Wr fused into wave 0's GEMM1 loop (lanes lr and
// lr+16 jointly hold row l0+lr's full 128 K-values across the 4 k-chunks,
// so one shfl_xor(16) completes the dot -> no separate pass over X).
// One block = 16 rows of L; 8 waves, wave w owns N-tiles {2w,2w+1} of both
// GEMMs. hmid staged through LDS as bf16 (stride 264 -> 16B-aligned,
// bank-spread). Weights come pre-packed; all fragment feeds are vector ops.
// Output stored transposed [B,256,L] (temporal: re-read by softmax from L2).
// ---------------------------------------------------------------------------
__global__ __launch_bounds__(256) void attn_kernel(const float* __restrict__ X,
                                                   const float* __restrict__ mask,
                                                   const __bf16* __restrict__ W1p,
                                                   const float* __restrict__ b1,
                                                   const __bf16* __restrict__ W2p,
                                                   const float* __restrict__ Wr,
                                                   float* __restrict__ attnT,
                                                   float* __restrict__ r) {
    __shared__ __bf16 hmid[16][KOUT + 8];   // stride 264 bf16 = 528 B

    const int tile = blockIdx.x;            // 0 .. B*L/16 - 1
    const int b    = (tile * 16) / LSEQ;
    const int l0   = (tile * 16) % LSEQ;
    const int wave = threadIdx.x >> 5;
    const int lane = threadIdx.x & 31;
    const int half = lane >> 4;
    const int lr   = lane & 15;

    // ---- GEMM1: X-tile[16x128] @ W1[128x256]  (+ fused X.Wr on wave 0) ----
    v8f acc0 = {}, acc1 = {};
    float rpart = 0.f;
    const float* xrow = X + (size_t)(b * LSEQ + l0 + lr) * DDIM;
    for (int kc = 0; kc < DDIM; kc += 32) {
        const int ak = kc + half * 8;
        const float4 q0 = *(const float4*)(xrow + ak);
        const float4 q1 = *(const float4*)(xrow + ak + 4);
        const float4 q2 = *(const float4*)(xrow + ak + 16);
        const float4 q3 = *(const float4*)(xrow + ak + 20);
        if (wave == 0) {                    // fused r projection (wave-uniform branch)
            const float4 w0 = *(const float4*)(Wr + ak);
            const float4 w1 = *(const float4*)(Wr + ak + 4);
            const float4 w2 = *(const float4*)(Wr + ak + 16);
            const float4 w3 = *(const float4*)(Wr + ak + 20);
            rpart += q0.x * w0.x + q0.y * w0.y + q0.z * w0.z + q0.w * w0.w
                   + q1.x * w1.x + q1.y * w1.y + q1.z * w1.z + q1.w * w1.w
                   + q2.x * w2.x + q2.y * w2.y + q2.z * w2.z + q2.w * w2.w
                   + q3.x * w3.x + q3.y * w3.y + q3.z * w3.z + q3.w * w3.w;
        }
        v16bf afrag;
        afrag[0]  = (__bf16)q0.x; afrag[1]  = (__bf16)q0.y;
        afrag[2]  = (__bf16)q0.z; afrag[3]  = (__bf16)q0.w;
        afrag[4]  = (__bf16)q1.x; afrag[5]  = (__bf16)q1.y;
        afrag[6]  = (__bf16)q1.z; afrag[7]  = (__bf16)q1.w;
        afrag[8]  = (__bf16)q2.x; afrag[9]  = (__bf16)q2.y;
        afrag[10] = (__bf16)q2.z; afrag[11] = (__bf16)q2.w;
        afrag[12] = (__bf16)q3.x; afrag[13] = (__bf16)q3.y;
        afrag[14] = (__bf16)q3.z; afrag[15] = (__bf16)q3.w;
        const __bf16* wb = W1p + ((size_t)((kc >> 5) * 16) * 32) * 16;
        v16bf b0 = *(const v16bf*)(wb + ((size_t)((wave * 2 + 0) * 32 + lane)) * 16);
        v16bf b1f = *(const v16bf*)(wb + ((size_t)((wave * 2 + 1) * 32 + lane)) * 16);
        acc0 = __builtin_amdgcn_wmma_f32_16x16x32_bf16(false, afrag, false, b0,
                                                       (short)0, acc0, false, false);
        acc1 = __builtin_amdgcn_wmma_f32_16x16x32_bf16(false, afrag, false, b1f,
                                                       (short)0, acc1, false, false);
    }
    if (wave == 0) {
        rpart += __shfl_xor(rpart, 16, 32);     // combine K-halves of row l0+lr
        if (lane < 16) r[b * LSEQ + l0 + lr] = rpart;
    }
    // bias + tanh -> bf16 LDS (C layout: M = v + 8*half, N = lr + 16*nt)
    #pragma unroll
    for (int t = 0; t < 2; ++t) {
        const int n    = (wave * 2 + t) * 16 + lr;
        const float bv = b1[n];
        const v8f  a   = t ? acc1 : acc0;
        #pragma unroll
        for (int v = 0; v < 8; ++v)
            hmid[v + half * 8][n] = (__bf16)fast_tanhf(a[v] + bv);
    }
    __syncthreads();

    // ---- GEMM2: hmid[16x256] @ W2[256x255] ----
    v8f c0 = {}, c1 = {};
    for (int kc = 0; kc < KOUT; kc += 32) {
        const int ak = kc + half * 8;
        const v8bf lo = *(const v8bf*)&hmid[lr][ak];        // 16B LDS vector loads
        const v8bf hi = *(const v8bf*)&hmid[lr][ak + 16];
        v16bf afrag;
        #pragma unroll
        for (int i = 0; i < 8; ++i) { afrag[i] = lo[i]; afrag[i + 8] = hi[i]; }
        const __bf16* wb = W2p + ((size_t)((kc >> 5) * 16) * 32) * 16;
        v16bf b0 = *(const v16bf*)(wb + ((size_t)((wave * 2 + 0) * 32 + lane)) * 16);
        v16bf b1f = *(const v16bf*)(wb + ((size_t)((wave * 2 + 1) * 32 + lane)) * 16);
        c0 = __builtin_amdgcn_wmma_f32_16x16x32_bf16(false, afrag, false, b0,
                                                     (short)0, c0, false, false);
        c1 = __builtin_amdgcn_wmma_f32_16x16x32_bf16(false, afrag, false, b1f,
                                                     (short)0, c1, false, false);
    }
    // mask-combine + transposed store: 8 C-elements per lane are consecutive
    // in L (M = v + 8*half) -> two 16B vector loads of mask, two 16B stores.
    const float4 m0 = *(const float4*)(mask + b * LSEQ + l0 + half * 8);
    const float4 m1 = *(const float4*)(mask + b * LSEQ + l0 + half * 8 + 4);
    const float mk[8] = { m0.x, m0.y, m0.z, m0.w, m1.x, m1.y, m1.z, m1.w };
    #pragma unroll
    for (int t = 0; t < 2; ++t) {
        const int n = (wave * 2 + t) * 16 + lr;
        const v8f c = t ? c1 : c0;
        float f[8];
        #pragma unroll
        for (int v = 0; v < 8; ++v)
            f[v] = mk[v] * c[v] + (1.f - mk[v]) * MASKV;
        float* dst = attnT + ((size_t)(b * KOUT + n)) * LSEQ + (l0 + half * 8);
        v4f s0 = { f[0], f[1], f[2], f[3] };
        v4f s1 = { f[4], f[5], f[6], f[7] };
        *(v4f*)dst       = s0;
        *(v4f*)(dst + 4) = s1;
    }
}

// ---------------------------------------------------------------------------
// Kernel 2: single-pass column softmax over L fused with mu = softmax . r + br.
// Each thread buffers its 16 column elements in registers (one global read of
// attnT total), then a 3-tuple (max, sumexp, dot) online-softmax tree combine.
// ---------------------------------------------------------------------------
__global__ __launch_bounds__(256) void softmax_mu_kernel(const float* __restrict__ attnT,
                                                         const float* __restrict__ r,
                                                         const float* __restrict__ br,
                                                         float* __restrict__ mu) {
    const int k = blockIdx.x;    // 0..254
    const int b = blockIdx.y;
    const float* col = attnT + ((size_t)(b * KOUT + k)) * LSEQ;
    const float* rb  = r + b * LSEQ;
    __shared__ float sm[256], ss[256], sd[256];
    const int t = threadIdx.x;

    float vbuf[16];
    #pragma unroll
    for (int i = 0; i < 16; ++i) vbuf[i] = col[t + i * 256];
    float ml = -INFINITY;
    #pragma unroll
    for (int i = 0; i < 16; ++i) ml = fmaxf(ml, vbuf[i]);
    float sel = 0.f, dotl = 0.f;
    #pragma unroll
    for (int i = 0; i < 16; ++i) {
        const float e = __expf(vbuf[i] - ml);
        sel  += e;
        dotl += e * rb[t + i * 256];
    }
    sm[t] = ml; ss[t] = sel; sd[t] = dotl;
    __syncthreads();
    for (int s = 128; s > 0; s >>= 1) {
        if (t < s) {
            const float m1 = sm[t], m2 = sm[t + s];
            const float M  = fmaxf(m1, m2);
            const float e1 = __expf(m1 - M), e2 = __expf(m2 - M);
            sm[t] = M;
            ss[t] = ss[t] * e1 + ss[t + s] * e2;
            sd[t] = sd[t] * e1 + sd[t + s] * e2;
        }
        __syncthreads();
    }
    if (t == 0) mu[b * KOUT + k] = sd[0] / ss[0] + br[0];
}

// ---------------------------------------------------------------------------
// Kernel 3: modes = clip(cumsum(softmax([0, mu]))[:255]); precompute a,b.
// ---------------------------------------------------------------------------
__global__ __launch_bounds__(256) void modes_kernel(const float* __restrict__ mu,
                                                    float* __restrict__ mm,
                                                    float* __restrict__ aa,
                                                    float* __restrict__ bb) {
    const int b = blockIdx.x, t = threadIdx.x;
    __shared__ float sh[256];
    __shared__ float red[256];
    const float v = (t == 0) ? 0.f : mu[b * KOUT + (t - 1)];

    red[t] = v; __syncthreads();
    for (int s = 128; s > 0; s >>= 1) {
        if (t < s) red[t] = fmaxf(red[t], red[t + s]);
        __syncthreads();
    }
    const float mx = red[0]; __syncthreads();
    const float e = __expf(v - mx);
    red[t] = e; __syncthreads();
    for (int s = 128; s > 0; s >>= 1) {
        if (t < s) red[t] += red[t + s];
        __syncthreads();
    }
    const float p = e / red[0];

    sh[t] = p; __syncthreads();
    for (int off = 1; off < 256; off <<= 1) {       // inclusive Hillis-Steele scan
        const float add = (t >= off) ? sh[t - off] : 0.f;
        __syncthreads();
        sh[t] += add;
        __syncthreads();
    }
    if (t < KOUT - 1) {
        const float m  = fminf(fmaxf(sh[t], 1e-4f), 0.9999f);
        const float a  = fminf(fmaxf(m - WIDTH_F * 0.5f, 0.f), 1.f - WIDTH_F);
        const float bv = fminf(fmaxf(a + WIDTH_F, WIDTH_F), 1.f);
        mm[b * KOUT + t] = m;
        aa[b * KOUT + t] = a;
        bb[b * KOUT + t] = bv;
    }
}

// ---------------------------------------------------------------------------
// Kernel 4: gamma_scaled[b,l] = sum_k cdf (the /255 * 255 cancels), then
// almat[b,l,k] = max(1-|gs-k|,0). Writes are v4f non-temporal (final output,
// never re-read): 64 lanes x 16B cover a row, 4 rows in flight per iteration.
// ---------------------------------------------------------------------------
__global__ __launch_bounds__(256) void gamma_almat_kernel(const float* __restrict__ mm,
                                                          const float* __restrict__ aa,
                                                          const float* __restrict__ bb,
                                                          float* __restrict__ out) {
    const int b  = blockIdx.y;
    const int l0 = blockIdx.x * 256;
    const int t  = threadIdx.x;
    __shared__ float smm[KOUT - 1], saa[KOUT - 1], sbb[KOUT - 1];
    __shared__ float gs[256];
    if (t < KOUT - 1) {
        smm[t] = mm[b * KOUT + t];
        saa[t] = aa[b * KOUT + t];
        sbb[t] = bb[b * KOUT + t];
    }
    __syncthreads();

    const int l = l0 + t;
    const float x = (float)l * (1.0f / (float)(LSEQ - 1));
    float s = 0.f;
    for (int k = 0; k < KOUT - 1; ++k) {
        const float m = smm[k], a = saa[k], bv = sbb[k];
        const float inv_ba = 1.f / (bv - a);
        float c;
        if (x <= m) {
            float u = fminf(fmaxf((x - a) / (m - a), 0.f), 1.f);
            float u2 = u * u, u4 = u2 * u2, u8 = u4 * u4;
            c = (m - a) * inv_ba * (u8 * u8);
        } else {
            float u = fminf(fmaxf((bv - x) / (bv - m), 0.f), 1.f);
            float u2 = u * u, u4 = u2 * u2, u8 = u4 * u4;
            c = 1.f - (bv - m) * inv_ba * (u8 * u8);
        }
        s += c;
    }
    out[b * LSEQ + l] = s;          // gamma_scaled
    gs[t] = s;
    __syncthreads();

    float* almat = out + (size_t)BATCH * LSEQ + ((size_t)(b * LSEQ + l0)) * KOUT;
    const int rsel = t >> 6;            // 4 rows in flight
    const int c4   = (t & 63) * 4;      // 64 lanes x 4 columns = full row
    for (int i0 = 0; i0 < 256; i0 += 4) {
        const int row = i0 + rsel;
        const float g = gs[row];
        v4f val;
        val.x = fmaxf(1.f - fabsf(g - (float)(c4 + 0)), 0.f);
        val.y = fmaxf(1.f - fabsf(g - (float)(c4 + 1)), 0.f);
        val.z = fmaxf(1.f - fabsf(g - (float)(c4 + 2)), 0.f);
        val.w = fmaxf(1.f - fabsf(g - (float)(c4 + 3)), 0.f);
        __builtin_nontemporal_store(val, (v4f*)(almat + (size_t)row * KOUT + c4));
    }
}

// ---------------------------------------------------------------------------
extern "C" void kernel_launch(void* const* d_in, const int* in_sizes, int n_in,
                              void* d_out, int out_size, void* d_ws, size_t ws_size,
                              hipStream_t stream) {
    const float* X    = (const float*)d_in[0];   // [16,4096,128]
    const float* mask = (const float*)d_in[1];   // [16,4096]
    const float* W1   = (const float*)d_in[2];   // [128,256]
    const float* b1   = (const float*)d_in[3];   // [256]
    const float* W2   = (const float*)d_in[4];   // [256,255]
    const float* Wr   = (const float*)d_in[5];   // [128,1]
    const float* br   = (const float*)d_in[6];   // [1]
    float* out = (float*)d_out;                  // [B*L] gamma_scaled ++ [B*L*256] almat

    // Workspace: attnT + r + mu/mm/aa/bb (f32), then packed bf16 weights.
    float* attnT = (float*)d_ws;
    float* r     = attnT + (size_t)BATCH * KOUT * LSEQ;
    float* mu    = r  + (size_t)BATCH * LSEQ;
    float* mm    = mu + (size_t)BATCH * KOUT;
    float* aa    = mm + (size_t)BATCH * KOUT;
    float* bb    = aa + (size_t)BATCH * KOUT;
    __bf16* W1p  = (__bf16*)(bb + (size_t)BATCH * KOUT);   // 2048*16 bf16
    __bf16* W2p  = W1p + (size_t)2048 * 16;                // 4096*16 bf16

    pack_weights_kernel<<<24, 256, 0, stream>>>(W1, W2, W1p, W2p);
    attn_kernel<<<BATCH * LSEQ / 16, 256, 0, stream>>>(X, mask, W1p, b1, W2p, Wr,
                                                       attnT, r);
    dim3 g2(KOUT - 1, BATCH);
    softmax_mu_kernel<<<g2, 256, 0, stream>>>(attnT, r, br, mu);
    modes_kernel<<<BATCH, 256, 0, stream>>>(mu, mm, aa, bb);
    dim3 g4(LSEQ / 256, BATCH);
    gamma_almat_kernel<<<g4, 256, 0, stream>>>(mm, aa, bb, out);
}
